// DGCNN_cls_58557584114124
// MI455X (gfx1250) — compile-verified, matrix operand order checked by reading, not run on hardware
//
#include <hip/hip_runtime.h>

typedef __attribute__((ext_vector_type(16))) __bf16 v16bf;
typedef __attribute__((ext_vector_type(8)))  __bf16 v8bf;
typedef __attribute__((ext_vector_type(8)))  float  v8f;
typedef __attribute__((ext_vector_type(4)))  unsigned int u32x4;
typedef __attribute__((ext_vector_type(8)))  int  i32x8;
typedef __attribute__((ext_vector_type(4)))  int  i32x4;

#define BATCH 8
#define NPTS  4096
#define KNN   10
#define NEG_SLOPE 0.2f

#ifndef __has_builtin
#define __has_builtin(x) 0
#endif
#if __has_builtin(__builtin_amdgcn_tensor_load_to_lds) && \
    __has_builtin(__builtin_amdgcn_s_wait_tensorcnt)
#define CDNA5_HAS_TDM 1
#if __has_include(<hip/amd_detail/amd_gfx1250_TDM.h>)
#define CDNA5_TDM_6ARG 1          // therock-10.0 headers -> 6-arg builtin
#else
#define CDNA5_TDM_6ARG 0          // ROCm 7.2 -> 5-arg builtin
#endif
#else
#define CDNA5_HAS_TDM 0
#endif

// ---------------------------------------------------------------------------
// Load a 16x32-bf16 A/B WMMA fragment for this lane from a row of padded-K
// bf16 data.  Per the CDNA5 16-bit A-matrix layout, lane L (half = L/16)
// holds K = {8h..8h+7} in VGPRs 0-3 and K = {16+8h..16+8h+7} in VGPRs 4-7,
// i.e. two contiguous 16-byte chunks.
// ---------------------------------------------------------------------------
__device__ __forceinline__ v16bf load_frag_bf16(const __bf16* row, int kk, int h) {
  const v8bf* p0 = (const v8bf*)(row + kk * 32 + 8 * h);
  const v8bf* p1 = (const v8bf*)(row + kk * 32 + 16 + 8 * h);
  v8bf c0 = *p0;
  v8bf c1 = *p1;
  v16bf f;
#pragma unroll
  for (int i = 0; i < 8; ++i) { f[i] = c0[i]; f[8 + i] = c1[i]; }
  return f;
}

__device__ __forceinline__ float lrelu(float v) {
  return v >= 0.f ? v : NEG_SLOPE * v;
}

// Wave-local LDS drain: this wave's DS ops complete; wave32 lockstep makes the
// data visible to all lanes without a workgroup barrier.
__device__ __forceinline__ void wave_lds_fence() {
  asm volatile("s_wait_dscnt 0x0" ::: "memory");
}

#if CDNA5_HAS_TDM
// ---------------------------------------------------------------------------
// Tensor Data Mover: copy 16 rows x 64 bf16 (contiguous source rows) into LDS,
// inserting a 256-byte pad after every 128-byte row so the rows interleave
// into a [16][192]-bf16 tile.  D# fields per CDNA5 ISA:
//   group0: count=1 | lds_addr | global_addr | type=2
//   group1: data_size=2B, pad_enable, pad_interval=32 DW (enc 4),
//           pad_amount=64 DW (enc 63), tensor/tile dims 64x16, strides 64
// ---------------------------------------------------------------------------
__device__ __forceinline__ void tdm_copy_rows64_pad192(const __bf16* src, void* ldsDst) {
  unsigned long long ga = (unsigned long long)(size_t)src;
  unsigned lds = (unsigned)(size_t)ldsDst;   // low 32 bits = LDS offset
  u32x4 g0;
  g0[0] = 1u;                                            // count=1, user D#
  g0[1] = lds;                                           // lds_addr
  g0[2] = (unsigned)ga;                                  // global_addr[31:0]
  g0[3] = (unsigned)((ga >> 32) & 0x01FFFFFFu) | (2u << 30);  // addr[56:32]|type=2
  i32x8 g1;
  g1[0] = (int)((1u << 16) | (1u << 20) | (4u << 22) | (63u << 25));
  g1[1] = (int)(64u << 16);   // tensor_dim0 = 64 -> bits[79:48] low half
  g1[2] = (int)(16u << 16);   // tensor_dim1 = 16 -> bits[111:80] low half
  g1[3] = (int)(64u << 16);   // tile_dim0  = 64 -> bits[127:112]
  g1[4] = (int)16;            // tile_dim1  = 16; tile_dim2 = 0
  g1[5] = 64;                 // tensor_dim0_stride = 64 (low 32)
  g1[6] = (int)(64u << 16);   // dim0_stride hi=0; tensor_dim1_stride low16 = 64
  g1[7] = 0;
  i32x4 gz = {0, 0, 0, 0};
#if CDNA5_TDM_6ARG
  i32x8 gz8 = {0, 0, 0, 0, 0, 0, 0, 0};
  __builtin_amdgcn_tensor_load_to_lds(g0, g1, gz, gz, gz8, 0);
#else
  __builtin_amdgcn_tensor_load_to_lds(g0, g1, gz, gz, 0);
#endif
}
#endif  // CDNA5_HAS_TDM

// ---------------------------------------------------------------------------
// Transpose (B,C,N) f32 -> (B,N,CP) bf16 (zero padded), plus squared norms.
// ---------------------------------------------------------------------------
__global__ void prep_feat_kernel(const float* __restrict__ src, int Cuse, int Cb, int CP,
                                 __bf16* __restrict__ xT, float* __restrict__ xx) {
  int tID = blockIdx.x * blockDim.x + threadIdx.x;
  if (tID >= BATCH * NPTS) return;
  int b = tID / NPTS, n = tID % NPTS;
  const float* sp = src + (size_t)b * Cb * NPTS + n;
  __bf16* op = xT + (size_t)tID * CP;
  float ss = 0.f;
  for (int c = 0; c < CP; ++c) {
    float v = (c < Cuse) ? sp[(size_t)c * NPTS] : 0.f;
    ss += v * v;
    op[c] = (__bf16)v;
  }
  xx[tID] = ss;
}

// Convert a weight matrix (O x Cin) f32 -> (O x CP) bf16, zero padded in K.
__global__ void prep_w_kernel(const float* __restrict__ W, int O, int Cin, int CP,
                              __bf16* __restrict__ Wp) {
  int o = blockIdx.x * blockDim.x + threadIdx.x;
  if (o >= O) return;
  for (int c = 0; c < CP; ++c)
    Wp[(size_t)o * CP + c] = (__bf16)((c < Cin) ? W[(size_t)o * Cin + c] : 0.f);
}

// ---------------------------------------------------------------------------
// Streaming KNN top-10: per wave 16 rows, loop 16-column tiles.
// Gram via bf16 WMMA, pd = 2*gram - ||xi||^2 - ||xj||^2, register top-k.
// dtile is wave-private, so only a wave-level DS drain is needed (no barrier).
// ---------------------------------------------------------------------------
template<int CP>
__global__ __launch_bounds__(128) void knn_topk_kernel(const __bf16* __restrict__ xT,
                                                       const float* __restrict__ xx,
                                                       int* __restrict__ idx) {
  constexpr int KK = CP / 32;
  __shared__ float dtile[4][16 * 16];

  const int b = blockIdx.y;
  const int lane = threadIdx.x & 31;
  const int w = threadIdx.x >> 5;
  const int h = lane >> 4;
  const int m = lane & 15;
  const int rowBase = blockIdx.x * 64 + w * 16;

  const __bf16* arow = xT + ((size_t)b * NPTS + rowBase + m) * CP;
  v16bf afr[KK];
#pragma unroll
  for (int kk = 0; kk < KK; ++kk) afr[kk] = load_frag_bf16(arow, kk, h);

  float xxr[8];
#pragma unroll
  for (int r = 0; r < 8; ++r) xxr[r] = xx[b * NPTS + rowBase + r + 8 * h];

  float tv[KNN];
  int   ti[KNN];
#pragma unroll
  for (int q = 0; q < KNN; ++q) { tv[q] = -3.0e38f; ti[q] = 0; }

  for (int ct = 0; ct < NPTS / 16; ++ct) {
    const int colBase = ct * 16;
    const __bf16* brow = xT + ((size_t)b * NPTS + colBase + m) * CP;
    __builtin_prefetch(brow + 16 * CP, 0, 3);   // next tile, WGP-resident hint
    v8f acc = {};
#pragma unroll
    for (int kk = 0; kk < KK; ++kk) {
      v16bf bfr = load_frag_bf16(brow, kk, h);
      acc = __builtin_amdgcn_wmma_f32_16x16x32_bf16(false, afr[kk], false, bfr,
                                                    (short)0, acc, false, false);
    }
    const float xxc = xx[b * NPTS + colBase + m];
#pragma unroll
    for (int r = 0; r < 8; ++r)
      dtile[w][(r + 8 * h) * 16 + m] = 2.0f * acc[r] - xxr[r] - xxc;
    wave_lds_fence();

    if (lane < 16) {
#pragma unroll
      for (int j = 0; j < 16; ++j) {
        float cand = dtile[w][lane * 16 + j];
        int cid = colBase + j;
        if (cand > tv[KNN - 1]) {
          tv[KNN - 1] = cand; ti[KNN - 1] = cid;
#pragma unroll
          for (int q = KNN - 1; q > 0; --q) {
            if (tv[q] > tv[q - 1]) {
              float tf = tv[q]; tv[q] = tv[q - 1]; tv[q - 1] = tf;
              int   tg = ti[q]; ti[q] = ti[q - 1]; ti[q - 1] = tg;
            }
          }
        }
      }
    }
    wave_lds_fence();   // scan loads complete before next iteration's stores
  }

  if (lane < 16) {
    int row = rowBase + lane;
#pragma unroll
    for (int q = 0; q < KNN; ++q)
      idx[((size_t)b * NPTS + row) * KNN + q] = ti[q];
  }
}

// ---------------------------------------------------------------------------
// Fused edge-conv stage:  gather -> edge features (LDS, bf16) ->
// WMMA GEMM layer(s) with scale/shift/LeakyReLU -> mean over K neighbors.
// Block: 128 threads, 8 points (80 edge columns).
// EM==1: 14-ch features from raw x (7ch, strided f32).
// EM==2: 128-ch features gathered from the bf16-transposed (B,N,64) rows
//        (two contiguous 128B rows per edge instead of 128 strided loads).
// ---------------------------------------------------------------------------
template<int CP1, int EM, bool TWO>
__global__ __launch_bounds__(128) void edgeconv_kernel(
    const float* __restrict__ xsrc,          // EM==1 only (B,7,N) f32
    const __bf16* __restrict__ xTsrc,        // EM==2 only (B,N,64) bf16
    const int* __restrict__ idx,
    const __bf16* __restrict__ Wa, const float* __restrict__ sa, const float* __restrict__ ta,
    const __bf16* __restrict__ Wb, const float* __restrict__ sb, const float* __restrict__ tb,
    float* __restrict__ xout)
{
  constexpr int COLS = 80;           // 8 points x 10 neighbors
  constexpr int KK1 = CP1 / 32;
  __shared__ __attribute__((aligned(16))) __bf16 F1[COLS * CP1];
  __shared__ __attribute__((aligned(16))) __bf16 F2[COLS * 64];
  __shared__ float OUTT[64 * COLS];

  const int b = blockIdx.y;
  const int nBase = blockIdx.x * 8;
  const int tid = threadIdx.x;
  const int lane = tid & 31;
  const int w = tid >> 5;
  const int h = lane >> 4;
  const int m = lane & 15;

  // ---- build gathered edge features in LDS ----
  for (int c = tid; c < COLS; c += 128) {
    int n = nBase + c / KNN;
    int k = c - (c / KNN) * KNN;
    int j = idx[((size_t)b * NPTS + n) * KNN + k];
    __bf16* f = &F1[c * CP1];
    if (EM == 1) {
      const size_t bb = (size_t)b * 7 * NPTS;
      float fv[7], cv[7];
#pragma unroll
      for (int ch = 0; ch < 7; ++ch) {
        fv[ch] = xsrc[bb + (size_t)ch * NPTS + j];
        cv[ch] = xsrc[bb + (size_t)ch * NPTS + n];
      }
      f[0] = (__bf16)(fv[0] - cv[0]);
      f[1] = (__bf16)(fv[1] - cv[1]);
#pragma unroll
      for (int ch = 2; ch < 7; ++ch) f[ch] = (__bf16)fv[ch];
#pragma unroll
      for (int ch = 0; ch < 7; ++ch) f[7 + ch] = (__bf16)cv[ch];
      for (int ch = 14; ch < CP1; ++ch) f[ch] = (__bf16)0.0f;
    } else {
      const __bf16* frow = xTsrc + ((size_t)b * NPTS + j) * 64;
      const __bf16* crow = xTsrc + ((size_t)b * NPTS + n) * 64;
#pragma unroll 8
      for (int ch = 0; ch < 64; ++ch) {
        float fv = (float)frow[ch];
        float cv = (float)crow[ch];
        f[ch]      = (__bf16)(fv - cv);
        f[64 + ch] = crow[ch];
      }
    }
  }
  __syncthreads();

  // ---- layer 1: (64 x CP1) x (CP1 x 80), wave w owns output rows w*16.. ----
  {
    const __bf16* arow = Wa + (size_t)(w * 16 + m) * CP1;
    v16bf afr[KK1];
#pragma unroll
    for (int kk = 0; kk < KK1; ++kk) afr[kk] = load_frag_bf16(arow, kk, h);
    for (int ctile = 0; ctile < COLS / 16; ++ctile) {
      v8f acc = {};
#pragma unroll
      for (int kk = 0; kk < KK1; ++kk) {
        v16bf bfr = load_frag_bf16(&F1[(ctile * 16 + m) * CP1], kk, h);
        acc = __builtin_amdgcn_wmma_f32_16x16x32_bf16(false, afr[kk], false, bfr,
                                                      (short)0, acc, false, false);
      }
#pragma unroll
      for (int r = 0; r < 8; ++r) {
        int o = w * 16 + r + 8 * h;
        int col = ctile * 16 + m;
        float v = lrelu(acc[r] * sa[o] + ta[o]);
        if (TWO) F2[col * 64 + o] = (__bf16)v;
        else     OUTT[o * COLS + col] = v;
      }
    }
  }
  __syncthreads();

  // ---- layer 2 (optional): (64 x 64) x (64 x 80) ----
  if (TWO) {
    const __bf16* arow = Wb + (size_t)(w * 16 + m) * 64;
    v16bf afr[2];
#pragma unroll
    for (int kk = 0; kk < 2; ++kk) afr[kk] = load_frag_bf16(arow, kk, h);
    for (int ctile = 0; ctile < COLS / 16; ++ctile) {
      v8f acc = {};
#pragma unroll
      for (int kk = 0; kk < 2; ++kk) {
        v16bf bfr = load_frag_bf16(&F2[(ctile * 16 + m) * 64], kk, h);
        acc = __builtin_amdgcn_wmma_f32_16x16x32_bf16(false, afr[kk], false, bfr,
                                                      (short)0, acc, false, false);
      }
#pragma unroll
      for (int r = 0; r < 8; ++r) {
        int o = w * 16 + r + 8 * h;
        int col = ctile * 16 + m;
        OUTT[o * COLS + col] = lrelu(acc[r] * sb[o] + tb[o]);
      }
    }
    __syncthreads();
  }

  // ---- mean over K neighbors ----
  for (int t = tid; t < 64 * 8; t += 128) {
    int o = t >> 3;
    int n = t & 7;
    float s = 0.f;
#pragma unroll
    for (int k = 0; k < KNN; ++k) s += OUTT[o * COLS + n * KNN + k];
    xout[((size_t)b * 64 + o) * NPTS + nBase + n] = s * (1.0f / KNN);
  }
}

// ---------------------------------------------------------------------------
// Pointwise conv (cbr1) via WMMA; the concatenated input channels come from
// the bf16-transposed (B,N,64) feature rows.
// MODE 0: C=192 concat(x1,x2,x3) -> h6.  The three interleaved row copies are
//         issued as TDM tensor_load_to_lds ops with LDS padding when the
//         toolchain exposes the builtin (manual copy fallback otherwise).
// MODE 1: C=448 concat(g,x1,x2,x3) -> h7 (gap of 192 DWORDs exceeds the
//         pad_amount field, so this mode always uses the manual copy).
// Block: 128 threads, 16 points.
// ---------------------------------------------------------------------------
template<int MODE>
__global__ __launch_bounds__(128) void cbr1_wmma_kernel(
    const __bf16* __restrict__ Wp, int O,
    const __bf16* __restrict__ xT1, const __bf16* __restrict__ xT2,
    const __bf16* __restrict__ xT3,
    const float* __restrict__ g,
    const float* __restrict__ s, const float* __restrict__ t,
    float* __restrict__ out)
{
  constexpr int CP = MODE ? 448 : 192;
  constexpr int GOFF = MODE ? 256 : 0;   // channels taken from g first
  __shared__ __attribute__((aligned(16))) __bf16 F[16 * CP];

  const int b = blockIdx.y;
  const int nBase = blockIdx.x * 16;
  const int tid = threadIdx.x;
  const int lane = tid & 31;
  const int w = tid >> 5;
  const int h = lane >> 4;
  const int m = lane & 15;

  auto build_manual = [&]() {
    for (int e = tid; e < 16 * CP; e += 128) {
      int n = nBase + e / CP;
      int k = e - (e / CP) * CP;
      __bf16 v;
      if (MODE == 1 && k < GOFF) {
        v = (__bf16)g[b * 256 + k];
      } else {
        int kk = k - GOFF;
        const __bf16* row;
        if (kk < 64)       row = xT1 + ((size_t)b * NPTS + n) * 64 + kk;
        else if (kk < 128) row = xT2 + ((size_t)b * NPTS + n) * 64 + (kk - 64);
        else               row = xT3 + ((size_t)b * NPTS + n) * 64 + (kk - 128);
        v = *row;
      }
      F[e] = v;
    }
  };

#if CDNA5_HAS_TDM
  if (MODE == 0) {
    if (tid < 32) {   // wave 0 issues the three DMA tile loads
      const size_t rowOff = ((size_t)b * NPTS + nBase) * 64;
      tdm_copy_rows64_pad192(xT1 + rowOff, (void*)(F + 0));
      tdm_copy_rows64_pad192(xT2 + rowOff, (void*)(F + 64));
      tdm_copy_rows64_pad192(xT3 + rowOff, (void*)(F + 128));
      __builtin_amdgcn_s_wait_tensorcnt(0);
    }
  } else {
    build_manual();
  }
#else
  build_manual();
#endif
  __syncthreads();

  for (int rt = w; rt < O / 16; rt += 4) {
    const __bf16* arow = Wp + (size_t)(rt * 16 + m) * CP;
    v8f acc = {};
    for (int kk = 0; kk < CP / 32; ++kk) {
      v16bf afr = load_frag_bf16(arow, kk, h);
      v16bf bfr = load_frag_bf16(&F[m * CP], kk, h);
      acc = __builtin_amdgcn_wmma_f32_16x16x32_bf16(false, afr, false, bfr,
                                                    (short)0, acc, false, false);
    }
#pragma unroll
    for (int r = 0; r < 8; ++r) {
      int o = rt * 16 + r + 8 * h;
      out[((size_t)b * O + o) * NPTS + nBase + m] = lrelu(acc[r] * s[o] + t[o]);
    }
  }
}

// ---------------------------------------------------------------------------
// Reductions + head (float4-vectorized over the contiguous N axis).
// ---------------------------------------------------------------------------
__global__ void mean_n_kernel(const float* __restrict__ hbuf, float* __restrict__ g, int O) {
  int tID = blockIdx.x * blockDim.x + threadIdx.x;
  if (tID >= BATCH * O) return;
  const float4* p = (const float4*)(hbuf + (size_t)tID * NPTS);
  float s = 0.f;
  for (int n = 0; n < NPTS / 4; ++n) {
    float4 v = p[n];
    s += (v.x + v.y) + (v.z + v.w);
  }
  g[tID] = s * (1.0f / NPTS);
}

__global__ void maxmean_n_kernel(const float* __restrict__ hbuf,
                                 float* __restrict__ x4, float* __restrict__ x5, int O) {
  int tID = blockIdx.x * blockDim.x + threadIdx.x;
  if (tID >= BATCH * O) return;
  const float4* p = (const float4*)(hbuf + (size_t)tID * NPTS);
  float s = 0.f, mx = -3.0e38f;
  for (int n = 0; n < NPTS / 4; ++n) {
    float4 v = p[n];
    s += (v.x + v.y) + (v.z + v.w);
    mx = fmaxf(mx, fmaxf(fmaxf(v.x, v.y), fmaxf(v.z, v.w)));
  }
  x4[tID] = mx;
  x5[tID] = s * (1.0f / NPTS);
}

__global__ void final_kernel(const float* __restrict__ x4, const float* __restrict__ x5,
                             const float* __restrict__ W8,
                             const float* __restrict__ s8, const float* __restrict__ t8,
                             float* __restrict__ out) {
  int tID = blockIdx.x * blockDim.x + threadIdx.x;
  if (tID >= BATCH * 128) return;
  int b = tID / 128, o = tID % 128;
  const float* wr = W8 + (size_t)o * 1024;
  float acc = 0.f;
  for (int c = 0; c < 512; ++c) acc += wr[c] * x4[b * 512 + c];
  for (int c = 0; c < 512; ++c) acc += wr[512 + c] * x5[b * 512 + c];
  out[tID] = lrelu(acc * s8[o] + t8[o]);
}

// ---------------------------------------------------------------------------
extern "C" void kernel_launch(void* const* d_in, const int* in_sizes, int n_in,
                              void* d_out, int out_size, void* d_ws, size_t ws_size,
                              hipStream_t stream) {
  (void)in_sizes; (void)n_in; (void)out_size; (void)ws_size;
  const float* x  = (const float*)d_in[0];
  const float* W1 = (const float*)d_in[1];
  const float* W2 = (const float*)d_in[2];
  const float* W3 = (const float*)d_in[3];
  const float* W4 = (const float*)d_in[4];
  const float* W5 = (const float*)d_in[5];
  const float* W6 = (const float*)d_in[6];
  const float* W7 = (const float*)d_in[7];
  const float* W8 = (const float*)d_in[8];
  // s/t are interleaved in dict order: s1,t1,s2,t2,...
  const float* s1 = (const float*)d_in[9];   const float* t1 = (const float*)d_in[10];
  const float* s2 = (const float*)d_in[11];  const float* t2 = (const float*)d_in[12];
  const float* s3 = (const float*)d_in[13];  const float* t3 = (const float*)d_in[14];
  const float* s4 = (const float*)d_in[15];  const float* t4 = (const float*)d_in[16];
  const float* s5 = (const float*)d_in[17];  const float* t5 = (const float*)d_in[18];
  const float* s6 = (const float*)d_in[19];  const float* t6 = (const float*)d_in[20];
  const float* s7 = (const float*)d_in[21];  const float* t7 = (const float*)d_in[22];
  const float* s8 = (const float*)d_in[23];  const float* t8 = (const float*)d_in[24];
  float* out = (float*)d_out;

  char* ws = (char*)d_ws;
  size_t off = 0;
  auto alloc = [&](size_t bytes) -> void* {
    void* p = ws + off;
    off += (bytes + 255) & ~(size_t)255;
    return p;
  };

  const size_t BN = (size_t)BATCH * NPTS;
  __bf16* xT2c  = (__bf16*)alloc(BN * 32 * 2);   // x[:, :2] padded to 32
  __bf16* xT64a = (__bf16*)alloc(BN * 64 * 2);   // x1 transposed bf16
  __bf16* xT64b = (__bf16*)alloc(BN * 64 * 2);   // x2 transposed bf16
  __bf16* xT64c = (__bf16*)alloc(BN * 64 * 2);   // x3 transposed bf16
  float* xx1 = (float*)alloc(BN * 4);
  float* xx2 = (float*)alloc(BN * 4);
  float* xx3 = (float*)alloc(BN * 4);
  int* idx1 = (int*)alloc(BN * KNN * 4);
  int* idx2 = (int*)alloc(BN * KNN * 4);
  int* idx3 = (int*)alloc(BN * KNN * 4);
  float* x1f = (float*)alloc(BN * 64 * 4);
  float* x2f = (float*)alloc(BN * 64 * 4);
  float* x3f = (float*)alloc(BN * 64 * 4);
  __bf16* Wp1 = (__bf16*)alloc(64 * 32 * 2);
  __bf16* Wp2 = (__bf16*)alloc(64 * 64 * 2);
  __bf16* Wp3 = (__bf16*)alloc(64 * 128 * 2);
  __bf16* Wp4 = (__bf16*)alloc(64 * 64 * 2);
  __bf16* Wp5 = (__bf16*)alloc(64 * 128 * 2);
  __bf16* Wp6 = (__bf16*)alloc(256 * 192 * 2);
  __bf16* Wp7 = (__bf16*)alloc(512 * 448 * 2);
  float* h6 = (float*)alloc(BN * 256 * 4);
  float* h7 = (float*)alloc(BN * 512 * 4);
  float* gbuf = (float*)alloc(BATCH * 256 * 4);
  float* x4b = (float*)alloc(BATCH * 512 * 4);
  float* x5b = (float*)alloc(BATCH * 512 * 4);

  // --- weight conversion (bf16, K padded) ---
  prep_w_kernel<<<1, 64, 0, stream>>>(W1, 64, 14, 32, Wp1);
  prep_w_kernel<<<1, 64, 0, stream>>>(W2, 64, 64, 64, Wp2);
  prep_w_kernel<<<1, 64, 0, stream>>>(W3, 64, 128, 128, Wp3);
  prep_w_kernel<<<1, 64, 0, stream>>>(W4, 64, 64, 64, Wp4);
  prep_w_kernel<<<1, 64, 0, stream>>>(W5, 64, 128, 128, Wp5);
  prep_w_kernel<<<2, 128, 0, stream>>>(W6, 256, 192, 192, Wp6);
  prep_w_kernel<<<4, 128, 0, stream>>>(W7, 512, 448, 448, Wp7);

  const int ptBlocks = (int)((BN + 127) / 128);

  // --- stage 1: knn on x[:, :2], edgeconv W1,W2 -> x1 ---
  prep_feat_kernel<<<ptBlocks, 128, 0, stream>>>(x, 2, 7, 32, xT2c, xx1);
  knn_topk_kernel<32><<<dim3(NPTS / 64, BATCH), 128, 0, stream>>>(xT2c, xx1, idx1);
  edgeconv_kernel<32, 1, true><<<dim3(NPTS / 8, BATCH), 128, 0, stream>>>(
      x, nullptr, idx1, Wp1, s1, t1, Wp2, s2, t2, x1f);

  // --- stage 2: knn on x1, edgeconv W3,W4 -> x2 ---
  prep_feat_kernel<<<ptBlocks, 128, 0, stream>>>(x1f, 64, 64, 64, xT64a, xx2);
  knn_topk_kernel<64><<<dim3(NPTS / 64, BATCH), 128, 0, stream>>>(xT64a, xx2, idx2);
  edgeconv_kernel<128, 2, true><<<dim3(NPTS / 8, BATCH), 128, 0, stream>>>(
      nullptr, xT64a, idx2, Wp3, s3, t3, Wp4, s4, t4, x2f);

  // --- stage 3: knn on x2, edgeconv W5 -> x3 ---
  prep_feat_kernel<<<ptBlocks, 128, 0, stream>>>(x2f, 64, 64, 64, xT64b, xx3);
  knn_topk_kernel<64><<<dim3(NPTS / 64, BATCH), 128, 0, stream>>>(xT64b, xx3, idx3);
  edgeconv_kernel<128, 2, false><<<dim3(NPTS / 8, BATCH), 128, 0, stream>>>(
      nullptr, xT64b, idx3, Wp5, s5, t5, (const __bf16*)nullptr, nullptr, nullptr, x3f);

  // --- bf16 transpose of x3 for the concat GEMMs (xx1 reused as scratch) ---
  prep_feat_kernel<<<ptBlocks, 128, 0, stream>>>(x3f, 64, 64, 64, xT64c, xx1);

  // --- tail: W6 -> h6, mean -> g, W7 -> h7, max/mean -> head ---
  cbr1_wmma_kernel<0><<<dim3(NPTS / 16, BATCH), 128, 0, stream>>>(
      Wp6, 256, xT64a, xT64b, xT64c, nullptr, s6, t6, h6);
  mean_n_kernel<<<(BATCH * 256 + 127) / 128, 128, 0, stream>>>(h6, gbuf, 256);
  cbr1_wmma_kernel<1><<<dim3(NPTS / 16, BATCH), 128, 0, stream>>>(
      Wp7, 512, xT64a, xT64b, xT64c, gbuf, s7, t7, h7);
  maxmean_n_kernel<<<(BATCH * 512 + 127) / 128, 128, 0, stream>>>(h7, x4b, x5b, 512);
  final_kernel<<<(BATCH * 128 + 127) / 128, 128, 0, stream>>>(x4b, x5b, W8, s8, t8, out);
}
